// MEAttention_7172595384820
// MI455X (gfx1250) — matrix operand order repaired and checked by reading, not compiled, not run on hardware
//
#include <hip/hip_runtime.h>
#include <hip/hip_bf16.h>

// MI455X / gfx1250, wave32. WMMA bf16 16x16x32 (fp32 acc) + TDM column gather.

typedef __attribute__((ext_vector_type(16))) __bf16 v16bf;
typedef __attribute__((ext_vector_type(8)))  __bf16 v8bf;
typedef __attribute__((ext_vector_type(8)))  float  v8f;
typedef __attribute__((ext_vector_type(4)))  unsigned int v4u;
typedef __attribute__((ext_vector_type(4)))  int    v4i;
typedef __attribute__((ext_vector_type(8)))  int    v8i;

#define BATCH  2
#define SQL    1024          // SQ == SK
#define DIMM   512
#define INNERD 64
#define ROWS   (BATCH * SQL) // 2048
#define NX     (ROWS * DIMM) // 1,048,576 activation elements
#define NW     (DIMM * INNERD) // 32,768 weight elements

// ---------------------------------------------------------------------------
// Kernel 0: precision prep.  bf16-convert activations; bf16-convert AND
// transpose weights to [N,K] so WMMA B-fragments become contiguous 32B loads.
// ---------------------------------------------------------------------------
__global__ __launch_bounds__(256) void prep_kernel(
    const float* __restrict__ x,  const float* __restrict__ emb,
    const float* __restrict__ Wq, const float* __restrict__ Wk,
    const float* __restrict__ Wv, const float* __restrict__ Wo,
    __bf16* __restrict__ xb,  __bf16* __restrict__ eb,
    __bf16* __restrict__ Wtq, __bf16* __restrict__ Wtk,
    __bf16* __restrict__ Wtv, __bf16* __restrict__ Wot)
{
    int idx = blockIdx.x * 256 + threadIdx.x;
    if (idx < NX) { xb[idx] = (__bf16)x[idx]; return; }
    idx -= NX;
    if (idx < NX) { eb[idx] = (__bf16)emb[idx]; return; }
    idx -= NX;
    if (idx < NW) { const int n = idx >> 9, k = idx & 511;           // [64][512]
                    Wtq[idx] = (__bf16)Wq[k * INNERD + n]; return; }
    idx -= NW;
    if (idx < NW) { const int n = idx >> 9, k = idx & 511;
                    Wtk[idx] = (__bf16)Wk[k * INNERD + n]; return; }
    idx -= NW;
    if (idx < NW) { const int n = idx >> 9, k = idx & 511;
                    Wtv[idx] = (__bf16)Wv[k * INNERD + n]; return; }
    idx -= NW;
    if (idx < NW) { const int n = idx >> 6, k = idx & 63;            // [512][64]
                    Wot[idx] = (__bf16)Wo[k * DIMM + n]; }
}

// ---------------------------------------------------------------------------
// Kernel 1: Q/K/V projections.  C[2048,64] = A[2048,512] x W[512,64], 3x.
// bf16 inputs; A fragment = 2x contiguous b128, B fragment = 1x v16bf load.
// ---------------------------------------------------------------------------
__global__ __launch_bounds__(256) void qkv_proj_kernel(
    const __bf16* __restrict__ xb,  const __bf16* __restrict__ eb,
    const __bf16* __restrict__ Wtq, const __bf16* __restrict__ Wtk,
    const __bf16* __restrict__ Wtv,
    float* __restrict__ Qo, float* __restrict__ Ko, float* __restrict__ Vo)
{
    const int lane = threadIdx.x & 31;
    const int wave = threadIdx.x >> 5;
    const int gtile = blockIdx.x * 8 + wave;        // 0..1535
    const int m  = gtile / 512;                     // which matrix (q/k/v)
    const int t  = gtile % 512;
    const int tM = t >> 2;                          // 0..127
    const int tN = t & 3;                           // 0..3

    const __bf16* A  = (m == 0) ? xb  : eb;
    const __bf16* Wt = (m == 0) ? Wtq : (m == 1 ? Wtk : Wtv);
    float*        C  = (m == 0) ? Qo  : (m == 1 ? Ko  : Vo);

    const int l15  = lane & 15;
    const int half = lane >> 4;                     // ISA lane-group (0/1)
    const __bf16* Arow = A  + (size_t)(tM * 16 + l15) * DIMM;
    const int col = tN * 16 + l15;
    const __bf16* Wrow = Wt + (size_t)col * DIMM;   // transposed: K contiguous

    v8f c = {0.f, 0.f, 0.f, 0.f, 0.f, 0.f, 0.f, 0.f};

    for (int kb = 0; kb < DIMM; kb += 32) {
        // A 16x32 bf16: K = half*16 + 0..15 (contiguous, matches B)
        const v8bf alo = *(const v8bf*)(Arow + kb + half * 16);
        const v8bf ahi = *(const v8bf*)(Arow + kb + half * 16 + 8);
        const v16bf a = __builtin_shufflevector(alo, ahi,
                          0,1,2,3,4,5,6,7,8,9,10,11,12,13,14,15);
        // B 32x16 bf16: col=lane%16, K = half*16 + 0..15 (contiguous in Wt)
        const v16bf b = *(const v16bf*)(Wrow + kb + half * 16);
        c = __builtin_amdgcn_wmma_f32_16x16x32_bf16(false, a, false, b,
                                                    (short)0, c, false, false);
    }

    // C/D 16x16 f32 layout: N = lane%16, M = (lane/16)*8 + r.
#pragma unroll
    for (int r = 0; r < 8; ++r) {
        const int mrow = tM * 16 + half * 8 + r;
        C[(size_t)mrow * INNERD + col] = c[r];
    }
}

// ---------------------------------------------------------------------------
// TDM column gather: tile 1 x 1024, element stride 64 floats, into LDS.
// D# packing per ISA 8.3 / 8.4 (count=1, type=2, data_size=4B).
// ---------------------------------------------------------------------------
__device__ __forceinline__ void tdm_load_column(const float* gptr,
                                                unsigned int lds_addr)
{
    const unsigned long long ga = (unsigned long long)(uintptr_t)gptr;
    v4u g0;
    g0[0] = 1u;                                    // count=1, user mode
    g0[1] = lds_addr;                              // lds_addr
    g0[2] = (unsigned int)ga;                      // global_addr[31:0]
    g0[3] = (unsigned int)(ga >> 32) | 0x80000000u; // addr[56:32] | type=2<<30
    v8i g1;
    g1[0] = 0x00020000;          // data_size=2 (4 bytes); no multicast/pad
    g1[1] = (int)(1u << 16);     // tensor_dim0 = 1     -> bits[63:48]
    g1[2] = (int)(1024u << 16);  // tensor_dim1 = 1024  -> bits[111:80]
    g1[3] = (int)(1u << 16);     // tile_dim0 = 1       -> bits[127:112]
    g1[4] = 1024;                // tile_dim1 = 1024; tile_dim2 = 0
    g1[5] = 64;                  // tensor_dim0_stride = 64 elements
    g1[6] = 0;
    g1[7] = 0;
    const v4i z4 = {0, 0, 0, 0};
#if defined(__clang_major__) && (__clang_major__ >= 23)
    const v8i z8 = {0, 0, 0, 0, 0, 0, 0, 0};
    __builtin_amdgcn_tensor_load_to_lds(g0, g1, z4, z4, z8, 0);
#else
    __builtin_amdgcn_tensor_load_to_lds(g0, g1, z4, z4, 0);
#endif
}

// ---------------------------------------------------------------------------
// Kernel 2: per-channel scalar attention (head-dim == 1).
// One block per (b, h); K/V channel columns gathered into LDS by the TDM,
// stable softmax via max(q*kmax, q*kmin) -- score matrix is rank-1.
// ---------------------------------------------------------------------------
__global__ __launch_bounds__(256) void attn_kernel(
    const float* __restrict__ Q, const float* __restrict__ Kp,
    const float* __restrict__ Vp, float* __restrict__ O)
{
    __shared__ float kcol[SQL];
    __shared__ float vcol[SQL];
    __shared__ float smax[8], smin[8];

    const int b   = blockIdx.x >> 6;   // /64
    const int h   = blockIdx.x & 63;
    const int tid = threadIdx.x;
    const int lane = tid & 31;
    const int wid  = tid >> 5;

    const size_t colbase = (size_t)b * SQL * INNERD + h;

    if (tid < 32) {                       // wave 0 drives the TDM
        tdm_load_column(Kp + colbase, (unsigned int)(uintptr_t)kcol);
        tdm_load_column(Vp + colbase, (unsigned int)(uintptr_t)vcol);
        __builtin_amdgcn_s_wait_tensorcnt(0);
    }
    __syncthreads();                      // LDS now holds both columns

    float lmax = -3.402823466e38f, lmin = 3.402823466e38f;
    for (int i = tid; i < SQL; i += 256) {
        const float kv = kcol[i];
        lmax = fmaxf(lmax, kv);
        lmin = fminf(lmin, kv);
    }
#pragma unroll
    for (int off = 16; off > 0; off >>= 1) {
        lmax = fmaxf(lmax, __shfl_xor(lmax, off, 32));
        lmin = fminf(lmin, __shfl_xor(lmin, off, 32));
    }
    if (lane == 0) { smax[wid] = lmax; smin[wid] = lmin; }
    __syncthreads();

    float kmax = smax[0], kmin = smin[0];
#pragma unroll
    for (int i = 1; i < 8; ++i) {
        kmax = fmaxf(kmax, smax[i]);
        kmin = fminf(kmin, smin[i]);
    }

    for (int q = tid; q < SQL; q += 256) {
        if (q + 256 < SQL)                // strided Q walk: prefetch next row
            __builtin_prefetch(&Q[colbase + (size_t)(q + 256) * INNERD], 0, 3);
        const float qv = Q[colbase + (size_t)q * INNERD];
        const float mlog = fmaxf(qv * kmax, qv * kmin);  // max_k(qv * k_k)
        float se = 0.f, sv = 0.f;
#pragma unroll 4
        for (int k = 0; k < SQL; ++k) {
            const float e = __expf(qv * kcol[k] - mlog);
            se += e;
            sv += e * vcol[k];
        }
        O[colbase + (size_t)q * INNERD] = sv / se;
    }
}

// ---------------------------------------------------------------------------
// Kernel 3: out[2048,512] = Oa[2048,64] x Wo[64,512] + bo.
// A (fp32 attention output) converted inline; B from transposed bf16 Wot.
// ---------------------------------------------------------------------------
__global__ __launch_bounds__(256) void out_proj_kernel(
    const float* __restrict__ Oa, const __bf16* __restrict__ Wot,
    const float* __restrict__ bo, float* __restrict__ out)
{
    const int lane = threadIdx.x & 31;
    const int wave = threadIdx.x >> 5;
    const int gtile = blockIdx.x * 8 + wave;    // 0..4095
    const int tM = gtile >> 5;                  // 0..127
    const int tN = gtile & 31;                  // 0..31

    const int l15  = lane & 15;
    const int half = lane >> 4;
    const float*  Arow = Oa  + (size_t)(tM * 16 + l15) * INNERD;
    const int col = tN * 16 + l15;
    const __bf16* Wrow = Wot + (size_t)col * INNERD;  // transposed [512][64]

    v8f c = {0.f, 0.f, 0.f, 0.f, 0.f, 0.f, 0.f, 0.f};

#pragma unroll
    for (int kb = 0; kb < INNERD; kb += 32) {
        v16bf a;
#pragma unroll
        for (int j = 0; j < 8; ++j) {
            const int ka = kb + half * 16 + 2 * j;
            const float2 f = *(const float2*)(Arow + ka);
            a[2 * j]     = (__bf16)f.x;
            a[2 * j + 1] = (__bf16)f.y;
        }
        const v16bf b = *(const v16bf*)(Wrow + kb + half * 16);
        c = __builtin_amdgcn_wmma_f32_16x16x32_bf16(false, a, false, b,
                                                    (short)0, c, false, false);
    }

    const float bias = bo[col];
#pragma unroll
    for (int r = 0; r < 8; ++r) {
        const int mrow = tM * 16 + half * 8 + r;
        out[(size_t)mrow * DIMM + col] = c[r] + bias;
    }
}

// ---------------------------------------------------------------------------
extern "C" void kernel_launch(void* const* d_in, const int* in_sizes, int n_in,
                              void* d_out, int out_size, void* d_ws, size_t ws_size,
                              hipStream_t stream)
{
    (void)in_sizes; (void)n_in; (void)out_size; (void)ws_size;

    const float* x   = (const float*)d_in[0];
    const float* emb = (const float*)d_in[1];
    const float* Wq  = (const float*)d_in[2];
    const float* Wk  = (const float*)d_in[3];
    const float* Wv  = (const float*)d_in[4];
    const float* Wo  = (const float*)d_in[5];
    const float* bo  = (const float*)d_in[6];
    float* out = (float*)d_out;

    // Workspace: f32 Q|K|V|o (2 MB) then bf16 xb|eb|Wtq|Wtk|Wtv|Wot (~4.25 MB)
    float* Qo = (float*)d_ws;
    float* Ko = Qo + (size_t)ROWS * INNERD;
    float* Vo = Ko + (size_t)ROWS * INNERD;
    float* Oa = Vo + (size_t)ROWS * INNERD;
    __bf16* xb  = (__bf16*)(Oa + (size_t)ROWS * INNERD);
    __bf16* eb  = xb  + (size_t)NX;
    __bf16* Wtq = eb  + (size_t)NX;
    __bf16* Wtk = Wtq + (size_t)NW;
    __bf16* Wtv = Wtk + (size_t)NW;
    __bf16* Wot = Wtv + (size_t)NW;

    const int prep_total = 2 * NX + 4 * NW;          // 2,228,224
    prep_kernel<<<(prep_total + 255) / 256, 256, 0, stream>>>(
        x, emb, Wq, Wk, Wv, Wo, xb, eb, Wtq, Wtk, Wtv, Wot);
    qkv_proj_kernel<<<192, 256, 0, stream>>>(xb, eb, Wtq, Wtk, Wtv, Qo, Ko, Vo);
    attn_kernel<<<BATCH * INNERD, 256, 0, stream>>>(Qo, Ko, Vo, Oa);
    out_proj_kernel<<<512, 256, 0, stream>>>(Oa, Wot, bo, out);
}
